// NATTENCrossAttentionBlock_16157666968363
// MI455X (gfx1250) — matrix operand order repaired and checked by reading, not compiled, not run on hardware
//
#include <hip/hip_runtime.h>
#include <hip/hip_bf16.h>

// ---------------------------------------------------------------------------
// CDNA5 (gfx1250) implementation of the NATTEN cross-attention block.
// Sizes (fixed): C=128, HK=WK=32, S=4 -> HQ=WQ=128, NH=4, HD=32, KH=KW=7, D=4.
// All GEMM-shaped work uses v_wmma_f32_16x16x32_f16 (f16 in, fp32 accum).
// ---------------------------------------------------------------------------

typedef __attribute__((ext_vector_type(16))) _Float16 f16x16;
typedef __attribute__((ext_vector_type(8)))  _Float16 f16x8;
typedef __attribute__((ext_vector_type(8)))  float    f32x8;

// Build a 16-half WMMA operand from two contiguous 8-half (16-byte) runs.
// A-matrix (16x32 f16): lane l holds row m=l&15; K run [kb..kb+15]
// with kb=(l>>4)*16 -> elems 0..15 contiguous (pass p1 = p0+8).
// B-matrix (32x16 f16): lane l holds col n=l&15; K run [(l>>4)*16 .. +15]
// -> elems 0..15 contiguous (pass p1 = p0+8).
static __device__ inline f16x16 frag2(const _Float16* p0, const _Float16* p1) {
  f16x8 lo = *(const f16x8*)p0;
  f16x8 hi = *(const f16x8*)p1;
  f16x16 r;
#pragma unroll
  for (int i = 0; i < 8; ++i) { r[i] = lo[i]; r[i + 8] = hi[i]; }
  return r;
}

static __device__ inline f32x8 wmma16(f16x16 a, f16x16 b, f32x8 c) {
  return __builtin_amdgcn_wmma_f32_16x16x32_f16(false, a, false, b,
                                                (short)0, c, false, false);
}

// ---------------------------------------------------------------------------
// Kernel 0: weight repack/convert.
// conv_w [oc][ic][3][3] f32  -> w16 [oc][j=ky*3+kx][ic] f16   (147456)
// qproj_w / kproj_w [o][c]   -> f16 copies                     (16384 each)
// ---------------------------------------------------------------------------
__global__ __launch_bounds__(256) void prep_weights(
    const float* __restrict__ conv_w, const float* __restrict__ qproj_w,
    const float* __restrict__ kproj_w, _Float16* __restrict__ w16,
    _Float16* __restrict__ qpw16, _Float16* __restrict__ kpw16) {
  int i = blockIdx.x * 256 + threadIdx.x;
  if (i < 147456) {
    int oc = i / 1152, r = i % 1152, j = r / 128, ic = r % 128;
    w16[i] = (_Float16)conv_w[(oc * 128 + ic) * 9 + j];
  }
  if (i < 16384) {
    qpw16[i] = (_Float16)qproj_w[i];
    kpw16[i] = (_Float16)kproj_w[i];
  }
}

// ---------------------------------------------------------------------------
// Kernel 1: 3x3 conv (implicit GEMM) + RMSNorm + 1x1 q-proj, fused.
// Grid: 256 blocks, each = 16(w) x 4(h) pixel tile, 256 threads = 8 waves.
// GEMM tiles: M=128 oc (8x16), N=64 px (4x16); wave w -> Mpair (w&3), Npair (w>>2).
// Softmax scale 1/sqrt(32) folded into stored q.
// ---------------------------------------------------------------------------
__global__ __launch_bounds__(256) void conv_q_kernel(
    const float* __restrict__ q, const _Float16* __restrict__ w16,
    const float* __restrict__ normq_w, const _Float16* __restrict__ qpw16,
    const float* __restrict__ qproj_b, _Float16* __restrict__ qp16) {
  __shared__ __align__(16) unsigned char smem[49152];
  __shared__ float rmsinv[64];
  __shared__ float partial[256];
  _Float16* qhalo = (_Float16*)smem;            // phase 1: [6][18][128] f16
  float*    qconv = (float*)smem;               // phase 2: [64 px][128 oc] f32
  _Float16* qn16  = (_Float16*)(smem + 32768);  // phase 2: [64 px][128 c] f16

  const int tid = threadIdx.x;
  const int bx = blockIdx.x & 7, by = blockIdx.x >> 3;
  const int x0 = bx * 16, y0 = by * 4;

  // Stage input halo (SAME padding) as f16, [row][col][ic] with ic contiguous.
  for (int idx = tid; idx < 6 * 18 * 128; idx += 256) {
    int ic = idx & 127, p = idx >> 7, hx = p % 18, hy = p / 18;
    int gy = y0 - 1 + hy, gx = x0 - 1 + hx;
    float val = 0.f;
    if (gy >= 0 && gy < 128 && gx >= 0 && gx < 128)
      val = q[ic * 16384 + gy * 128 + gx];
    qhalo[(hy * 18 + hx) * 128 + ic] = (_Float16)val;
  }
  __syncthreads();

  const int wv = tid >> 5, lane = tid & 31;
  const int mt0 = (wv & 3) * 2, nt0 = (wv >> 2) * 2;
  const int lm = lane & 15, lh = lane >> 4;
  const int kbA = lh * 16, kbB = lh * 16;

  f32x8 a00 = {}, a01 = {}, a10 = {}, a11 = {};
  for (int s = 0; s < 36; ++s) {          // K = 1152 = 36 steps of 32
    int j = s >> 2, ic0 = (s & 3) * 32;   // fixed 3x3 tap per step
    int ky = j / 3, kx = j % 3;
    const _Float16* a0p = w16 + ((mt0 * 16 + lm) * 9 + j) * 128 + ic0;
    const _Float16* a1p = w16 + (((mt0 + 1) * 16 + lm) * 9 + j) * 128 + ic0;
    f16x16 A0 = frag2(a0p + kbA, a0p + kbA + 8);
    f16x16 A1 = frag2(a1p + kbA, a1p + kbA + 8);
    const _Float16* b0p = qhalo + ((nt0 + ky) * 18 + lm + kx) * 128 + ic0 + kbB;
    const _Float16* b1p = qhalo + ((nt0 + 1 + ky) * 18 + lm + kx) * 128 + ic0 + kbB;
    f16x16 B0 = frag2(b0p, b0p + 8);
    f16x16 B1 = frag2(b1p, b1p + 8);
    a00 = wmma16(A0, B0, a00);
    a01 = wmma16(A0, B1, a01);
    a10 = wmma16(A1, B0, a10);
    a11 = wmma16(A1, B1, a11);
  }
  __syncthreads();  // all reads of qhalo done before overwrite

  {  // spill conv result to LDS as [px][oc]; D: m=r+8*lh, n=lm
    const int px0 = nt0 * 16 + lm, px1 = (nt0 + 1) * 16 + lm;
#pragma unroll
    for (int r = 0; r < 8; ++r) {
      int oc0 = mt0 * 16 + r + 8 * lh, oc1 = (mt0 + 1) * 16 + r + 8 * lh;
      qconv[px0 * 128 + oc0] = a00[r];
      qconv[px1 * 128 + oc0] = a01[r];
      qconv[px0 * 128 + oc1] = a10[r];
      qconv[px1 * 128 + oc1] = a11[r];
    }
  }
  __syncthreads();

  // RMS over channels, per pixel (4 partial sums per pixel).
  {
    int px = tid & 63, qd = tid >> 6;
    float ss = 0.f;
    for (int c = qd * 32; c < qd * 32 + 32; ++c) {
      float x = qconv[px * 128 + c];
      ss += x * x;
    }
    partial[qd * 64 + px] = ss;
  }
  __syncthreads();
  if (tid < 64) {
    float s = partial[tid] + partial[64 + tid] + partial[128 + tid] + partial[192 + tid];
    rmsinv[tid] = rsqrtf(s * (1.f / 128.f) + 1.1920929e-7f);
  }
  __syncthreads();
  for (int idx = tid; idx < 64 * 128; idx += 256) {
    int px = idx >> 7, c = idx & 127;
    qn16[idx] = (_Float16)(qconv[idx] * rmsinv[px] * normq_w[c]);
  }
  __syncthreads();

  // 1x1 q-projection GEMM, K=128.
  f32x8 p00 = {}, p01 = {}, p10 = {}, p11 = {};
  for (int s = 0; s < 4; ++s) {
    int ic0 = s * 32;
    const _Float16* a0p = qpw16 + (mt0 * 16 + lm) * 128 + ic0;
    const _Float16* a1p = qpw16 + ((mt0 + 1) * 16 + lm) * 128 + ic0;
    f16x16 A0 = frag2(a0p + kbA, a0p + kbA + 8);
    f16x16 A1 = frag2(a1p + kbA, a1p + kbA + 8);
    const _Float16* b0p = qn16 + (nt0 * 16 + lm) * 128 + ic0 + kbB;
    const _Float16* b1p = qn16 + ((nt0 + 1) * 16 + lm) * 128 + ic0 + kbB;
    f16x16 B0 = frag2(b0p, b0p + 8);
    f16x16 B1 = frag2(b1p, b1p + 8);
    p00 = wmma16(A0, B0, p00);
    p01 = wmma16(A0, B1, p01);
    p10 = wmma16(A1, B0, p10);
    p11 = wmma16(A1, B1, p11);
  }
  const float scl = 0.17677669529663687f;  // 1/sqrt(HD), folded into q
#pragma unroll
  for (int r = 0; r < 8; ++r) {
    int oc0 = mt0 * 16 + r + 8 * lh, oc1 = (mt0 + 1) * 16 + r + 8 * lh;
    int px0 = nt0 * 16 + lm, px1 = (nt0 + 1) * 16 + lm;
    int g0 = (y0 + (px0 >> 4)) * 128 + x0 + (px0 & 15);
    int g1 = (y0 + (px1 >> 4)) * 128 + x0 + (px1 & 15);
    qp16[g0 * 128 + oc0] = (_Float16)((p00[r] + qproj_b[oc0]) * scl);
    qp16[g1 * 128 + oc0] = (_Float16)((p01[r] + qproj_b[oc0]) * scl);
    qp16[g0 * 128 + oc1] = (_Float16)((p10[r] + qproj_b[oc1]) * scl);
    qp16[g1 * 128 + oc1] = (_Float16)((p11[r] + qproj_b[oc1]) * scl);
  }
}

// ---------------------------------------------------------------------------
// Kernel 2: RMSNorm + 1x1 k-proj over the 32x32 KV grid (16 blocks x 64 px).
// ---------------------------------------------------------------------------
__global__ __launch_bounds__(256) void k_kernel(
    const float* __restrict__ k, const float* __restrict__ normk_w,
    const _Float16* __restrict__ kpw16, const float* __restrict__ kproj_b,
    _Float16* __restrict__ kp16) {
  __shared__ __align__(16) float kf[64 * 128];
  __shared__ __align__(16) _Float16 kn16[64 * 128];
  __shared__ float rmsinv[64];
  __shared__ float partial[256];
  const int tid = threadIdx.x;
  const int p0 = blockIdx.x * 64;  // base KV pixel (2 rows of 32)

  for (int idx = tid; idx < 64 * 128; idx += 256) {
    int px = idx >> 7, c = idx & 127;
    kf[idx] = k[c * 1024 + p0 + px];
  }
  __syncthreads();
  {
    int px = tid & 63, qd = tid >> 6;
    float ss = 0.f;
    for (int c = qd * 32; c < qd * 32 + 32; ++c) {
      float x = kf[px * 128 + c];
      ss += x * x;
    }
    partial[qd * 64 + px] = ss;
  }
  __syncthreads();
  if (tid < 64) {
    float s = partial[tid] + partial[64 + tid] + partial[128 + tid] + partial[192 + tid];
    rmsinv[tid] = rsqrtf(s * (1.f / 128.f) + 1.1920929e-7f);
  }
  __syncthreads();
  for (int idx = tid; idx < 64 * 128; idx += 256) {
    int px = idx >> 7, c = idx & 127;
    kn16[idx] = (_Float16)(kf[idx] * rmsinv[px] * normk_w[c]);
  }
  __syncthreads();

  const int wv = tid >> 5, lane = tid & 31;
  const int mt0 = (wv & 3) * 2, nt0 = (wv >> 2) * 2;
  const int lm = lane & 15, lh = lane >> 4;
  const int kbA = lh * 16, kbB = lh * 16;

  f32x8 p00 = {}, p01 = {}, p10 = {}, p11 = {};
  for (int s = 0; s < 4; ++s) {
    int ic0 = s * 32;
    const _Float16* a0p = kpw16 + (mt0 * 16 + lm) * 128 + ic0;
    const _Float16* a1p = kpw16 + ((mt0 + 1) * 16 + lm) * 128 + ic0;
    f16x16 A0 = frag2(a0p + kbA, a0p + kbA + 8);
    f16x16 A1 = frag2(a1p + kbA, a1p + kbA + 8);
    const _Float16* b0p = kn16 + (nt0 * 16 + lm) * 128 + ic0 + kbB;
    const _Float16* b1p = kn16 + ((nt0 + 1) * 16 + lm) * 128 + ic0 + kbB;
    f16x16 B0 = frag2(b0p, b0p + 8);
    f16x16 B1 = frag2(b1p, b1p + 8);
    p00 = wmma16(A0, B0, p00);
    p01 = wmma16(A0, B1, p01);
    p10 = wmma16(A1, B0, p10);
    p11 = wmma16(A1, B1, p11);
  }
#pragma unroll
  for (int r = 0; r < 8; ++r) {
    int oc0 = mt0 * 16 + r + 8 * lh, oc1 = (mt0 + 1) * 16 + r + 8 * lh;
    int px0 = nt0 * 16 + lm, px1 = (nt0 + 1) * 16 + lm;
    kp16[(p0 + px0) * 128 + oc0] = (_Float16)(p00[r] + kproj_b[oc0]);
    kp16[(p0 + px1) * 128 + oc0] = (_Float16)(p01[r] + kproj_b[oc0]);
    kp16[(p0 + px0) * 128 + oc1] = (_Float16)(p10[r] + kproj_b[oc1]);
    kp16[(p0 + px1) * 128 + oc1] = (_Float16)(p11[r] + kproj_b[oc1]);
  }
}

// ---------------------------------------------------------------------------
// Kernel 3: NATTEN attention. One block = one 4x4 query block (16 queries),
// which shares a clamped 7x7 KV window: rs = clamp(by-3, 0, 25) (dilation-4
// window on the 4x-upsampled grid == plain clamped window in KV space).
// 128 threads = 4 waves: one head per wave for QK^T, fp32 softmax + head-mean,
// then attn(16x64) x V(64x128) via WMMA.
// ---------------------------------------------------------------------------
__global__ __launch_bounds__(128) void attn_kernel(
    const _Float16* __restrict__ qp16, const _Float16* __restrict__ kp16,
    const float* __restrict__ v, float* __restrict__ out) {
  __shared__ __align__(16) _Float16 kn[64 * 128];   // [key(pad64)][c]
  __shared__ __align__(16) _Float16 vT[128 * 64];   // [c][key(pad64)]
  __shared__ float lg[4 * 16 * 64];                 // [head][q][key]
  __shared__ __align__(16) _Float16 attnm[16 * 64]; // head-mean attention
  const int tid = threadIdx.x;
  const int bxk = blockIdx.x & 31, byk = blockIdx.x >> 5;
  int rs = byk - 3; if (rs < 0) rs = 0; if (rs > 25) rs = 25;
  int cs = bxk - 3; if (cs < 0) cs = 0; if (cs > 25) cs = 25;
  const int y0 = byk * 4, x0 = bxk * 4;

  for (int idx = tid; idx < 64 * 128; idx += 128) {  // keys for QK B-operand
    int key = idx >> 7, c = idx & 127;
    _Float16 val = (_Float16)0.f;
    if (key < 49) {
      int m = key / 7, n = key % 7;
      val = kp16[((rs + m) * 32 + cs + n) * 128 + c];
    }
    kn[idx] = val;
  }
  for (int idx = tid; idx < 128 * 64; idx += 128) {  // V for AV B-operand
    int c = idx >> 6, key = idx & 63;
    float vv = 0.f;
    if (key < 49) {
      int m = key / 7, n = key % 7;
      vv = v[c * 1024 + (rs + m) * 32 + cs + n];
    }
    vT[idx] = (_Float16)vv;
  }
  __syncthreads();

  const int wv = tid >> 5, lane = tid & 31;
  const int lm = lane & 15, lh = lane >> 4;
  const int kbA = lh * 16, kbB = lh * 16;

  {  // QK^T: head h = wave id; A = q block [16 q x 32 d]
    const int h = wv;
    const int gy = y0 + (lm >> 2), gx = x0 + (lm & 3);
    const _Float16* ap = qp16 + (gy * 128 + gx) * 128 + h * 32;
    f16x16 A = frag2(ap + kbA, ap + kbA + 8);
#pragma unroll
    for (int kt = 0; kt < 4; ++kt) {
      const _Float16* bp = kn + (kt * 16 + lm) * 128 + h * 32 + kbB;
      f16x16 B = frag2(bp, bp + 8);
      f32x8 d = {};
      d = wmma16(A, B, d);
#pragma unroll
      for (int r = 0; r < 8; ++r) {
        int qq = r + 8 * lh, key = kt * 16 + lm;
        lg[(h * 16 + qq) * 64 + key] = d[r];
      }
    }
  }
  __syncthreads();

  if (tid < 16) {  // softmax over 49 keys per head, then mean over heads
    int qq = tid;
    float mh[4], sh[4];
    for (int h = 0; h < 4; ++h) {
      float mx = -3.0e38f;
      for (int kk = 0; kk < 49; ++kk) mx = fmaxf(mx, lg[(h * 16 + qq) * 64 + kk]);
      float s = 0.f;
      for (int kk = 0; kk < 49; ++kk) s += __expf(lg[(h * 16 + qq) * 64 + kk] - mx);
      mh[h] = mx;
      sh[h] = 1.f / s;
    }
    for (int kk = 0; kk < 64; ++kk) {
      float a = 0.f;
      if (kk < 49) {
        for (int h = 0; h < 4; ++h)
          a += __expf(lg[(h * 16 + qq) * 64 + kk] - mh[h]) * sh[h];
        a *= 0.25f;
      }
      attnm[qq * 64 + kk] = (_Float16)a;
    }
  }
  __syncthreads();

  // attn (16 x 64) x V (64 x 128): 8 c-tiles, 2 per wave, K = 64 (2 steps).
  for (int cti = 0; cti < 2; ++cti) {
    const int ct = wv * 2 + cti;
    f32x8 d = {};
#pragma unroll
    for (int s = 0; s < 2; ++s) {
      const _Float16* ap = attnm + lm * 64 + s * 32;
      f16x16 A = frag2(ap + kbA, ap + kbA + 8);
      const _Float16* bp = vT + (ct * 16 + lm) * 64 + s * 32 + kbB;
      f16x16 B = frag2(bp, bp + 8);
      d = wmma16(A, B, d);
    }
#pragma unroll
    for (int r = 0; r < 8; ++r) {
      int qq = r + 8 * lh, c = ct * 16 + lm;
      int gy = y0 + (qq >> 2), gx = x0 + (qq & 3);
      out[c * 16384 + gy * 128 + gx] = d[r];
    }
  }
}

// ---------------------------------------------------------------------------
extern "C" void kernel_launch(void* const* d_in, const int* in_sizes, int n_in,
                              void* d_out, int out_size, void* d_ws, size_t ws_size,
                              hipStream_t stream) {
  const float* q       = (const float*)d_in[0];
  const float* k       = (const float*)d_in[1];
  const float* v       = (const float*)d_in[2];
  const float* conv_w  = (const float*)d_in[3];
  const float* normq_w = (const float*)d_in[4];
  const float* normk_w = (const float*)d_in[5];
  const float* qproj_w = (const float*)d_in[6];
  const float* qproj_b = (const float*)d_in[7];
  const float* kproj_w = (const float*)d_in[8];
  const float* kproj_b = (const float*)d_in[9];
  float* out = (float*)d_out;

  unsigned char* ws = (unsigned char*)d_ws;
  _Float16* w16   = (_Float16*)ws; ws += (size_t)147456 * 2;
  _Float16* qpw16 = (_Float16*)ws; ws += (size_t)16384 * 2;
  _Float16* kpw16 = (_Float16*)ws; ws += (size_t)16384 * 2;
  _Float16* qp16  = (_Float16*)ws; ws += (size_t)16384 * 128 * 2;
  _Float16* kp16  = (_Float16*)ws; ws += (size_t)1024 * 128 * 2;

  prep_weights<<<576, 256, 0, stream>>>(conv_w, qproj_w, kproj_w, w16, qpw16, kpw16);
  k_kernel<<<16, 256, 0, stream>>>(k, normk_w, kpw16, kproj_b, kp16);
  conv_q_kernel<<<256, 256, 0, stream>>>(q, w16, normq_w, qpw16, qproj_b, qp16);
  attn_kernel<<<1024, 128, 0, stream>>>(qp16, kp16, v, out);
}
